// SparseDispatcher_65257733096084
// MI455X (gfx1250) — matrix operand order repaired and checked by reading, not compiled, not run on hardware
//
#include <hip/hip_runtime.h>
#include <hip/hip_bf16.h>

// MoE log(sum_e gate*exp(inp@W[e]+b[e])), dense over E=8 experts.
// Pipeline: (1) one-time f32->bf16 conversion of inp, and transpose+convert of
// W into [e][n][k] bf16 (both land in d_ws, L2-resident on a 192MB-L2 part),
// (2) main GEMM kernel: v_wmma_f32_16x16x32_bf16, 128x128 block tile, 8 wave32
// waves (4x2), double-buffered LDS stages filled by GLOBAL_LOAD_ASYNC_TO_LDS
// (ASYNCcnt-tracked) overlapping the WMMA stream.

typedef __attribute__((ext_vector_type(16))) __bf16 v16bf;
typedef __attribute__((ext_vector_type(8)))  __bf16 v8bf;
typedef __attribute__((ext_vector_type(8)))  float  v8f;
typedef __attribute__((ext_vector_type(4)))  int    v4i;

#define NEXP 8
#define DIN  512
#define DOUT 512
#define BM   128
#define BN   128
#define BK   64
#define LDT  72   // stage row stride (64 + 8 pad): 144B, 16B-aligned, bank-spread

#if __has_builtin(__builtin_amdgcn_global_load_async_to_lds_b128)
#define HAVE_ASYNC_LDS 1
#endif
#if __has_builtin(__builtin_amdgcn_s_wait_asynccnt)
#define WAIT_ASYNC0() __builtin_amdgcn_s_wait_asynccnt(0)
#else
#define WAIT_ASYNC0() asm volatile("s_wait_asynccnt 0" ::: "memory")
#endif

typedef __attribute__((address_space(1))) v4i as1_v4i;   // global
typedef __attribute__((address_space(3))) v4i as3_v4i;   // LDS

static __device__ __forceinline__ void cp16(const __bf16* g, __bf16* l) {
#ifdef HAVE_ASYNC_LDS
    __builtin_amdgcn_global_load_async_to_lds_b128(
        (as1_v4i*)(void*)g, (as3_v4i*)(void*)l, 0, 0);
#else
    *(v8bf*)l = *(const v8bf*)g;
#endif
}

// ---------------- pre-pass 1: inp f32 -> bf16 (elementwise) ----------------
__global__ __launch_bounds__(256)
void cvt_inp_kernel(const float* __restrict__ in, __bf16* __restrict__ out) {
    size_t t = (size_t)blockIdx.x * 256 + threadIdx.x;   // one v8bf per thread
    const float4* p = (const float4*)in + t * 2;
    float4 a = p[0], b = p[1];
    v8bf o;
    o[0] = (__bf16)a.x; o[1] = (__bf16)a.y; o[2] = (__bf16)a.z; o[3] = (__bf16)a.w;
    o[4] = (__bf16)b.x; o[5] = (__bf16)b.y; o[6] = (__bf16)b.z; o[7] = (__bf16)b.w;
    *((v8bf*)out + t) = o;
}

// ------- pre-pass 2: W [e][k][n] f32 -> Wt [e][n][k] bf16 (LDS tiled) -------
__global__ __launch_bounds__(256)
void cvt_w_transpose_kernel(const float* __restrict__ W, __bf16* __restrict__ Wt) {
    __shared__ float T[64 * 65];
    const int tid = threadIdx.x;
    const int e  = blockIdx.x >> 6;            // 8 experts
    const int tI = blockIdx.x & 63;            // 8x8 tiles of 64x64
    const int tk = (tI >> 3) * 64;
    const int tn = (tI & 7) * 64;
    // load 64(k) x 64(n) f32 tile, coalesced along n
#pragma unroll
    for (int i = 0; i < 4; ++i) {
        int linear = tid + i * 256;            // 0..1023 float4
        int r  = linear >> 4;
        int c4 = (linear & 15) << 2;
        float4 v = *(const float4*)(&W[((size_t)e * DIN + tk + r) * DOUT + tn + c4]);
        T[r * 65 + c4 + 0] = v.x; T[r * 65 + c4 + 1] = v.y;
        T[r * 65 + c4 + 2] = v.z; T[r * 65 + c4 + 3] = v.w;
    }
    __syncthreads();
    // store transposed bf16, coalesced along k
#pragma unroll
    for (int i = 0; i < 2; ++i) {
        int linear = tid + i * 256;            // 0..511 chunks of 8k
        int n_l = linear >> 3;
        int kc  = (linear & 7) << 3;
        v8bf o;
#pragma unroll
        for (int q = 0; q < 8; ++q) o[q] = (__bf16)T[(kc + q) * 65 + n_l];
        *(v8bf*)(&Wt[((size_t)e * DOUT + tn + n_l) * DIN + tk + kc]) = o;
    }
}

// ----------------------------- main GEMM kernel -----------------------------
__global__ __launch_bounds__(256)
void moe_logexp_wmma_bf16(const float*  __restrict__ gates,   // [B, 8]
                          const __bf16* __restrict__ inp_bf,  // [B, 512]
                          const __bf16* __restrict__ Wt,      // [8, 512(n), 512(k)]
                          const float*  __restrict__ bias,    // [8, 512]
                          float* __restrict__ out)            // [B, 512]
{
    __shared__ __align__(16) __bf16 A2[2][BM * LDT];   // [row][k] stage, 2x18KB
    __shared__ __align__(16) __bf16 B2[2][BN * LDT];   // [n][k]  stage, 2x18KB

    const int tid  = threadIdx.x;
    const int lane = tid & 31;          // wave32
    const int wv   = tid >> 5;          // 0..7
    const int wm   = wv >> 1;           // 0..3 : 32-row strip
    const int wn   = wv & 1;            // 0..1 : 64-col strip
    const int ln15 = lane & 15;
    const int hi   = lane >> 4;

    const int rowBase = (blockIdx.x >> 2) * BM;   // 256 row blocks
    const int colBase = (blockIdx.x & 3) * BN;    // 4 col blocks

    // stage issue: copy A (inp rows x 64k) and B (Wt rows x 64k) bf16 tiles
    // into LDS buffer p. 1024 16B chunks each / 256 threads = 4+4 per thread.
    auto issueStage = [&](int e, int kk, int p) {
        const __bf16* asrc = inp_bf + (size_t)rowBase * DIN + kk;
        const __bf16* bsrc = Wt + ((size_t)e * DOUT + colBase) * DIN + kk;
        __bf16* adst = &A2[p][0];
        __bf16* bdst = &B2[p][0];
#pragma unroll
        for (int i = 0; i < 4; ++i) {
            int c  = tid + i * 256;        // 0..1023
            int r  = c >> 3;               // 0..127 row
            int ko = (c & 7) << 3;         // 0..56 (elements)
            cp16(asrc + (size_t)r * DIN + ko, adst + r * LDT + ko);
            cp16(bsrc + (size_t)r * DIN + ko, bdst + r * LDT + ko);
        }
    };

    v8f comb[2][4];
#pragma unroll
    for (int mt = 0; mt < 2; ++mt)
#pragma unroll
        for (int nt = 0; nt < 4; ++nt)
#pragma unroll
            for (int j = 0; j < 8; ++j)
                comb[mt][nt][j] = 0.0f;

    issueStage(0, 0, 0);   // prologue
    int s = 0;

    for (int e = 0; e < NEXP; ++e) {
        v8f acc[2][4];
#pragma unroll
        for (int mt = 0; mt < 2; ++mt)
#pragma unroll
            for (int nt = 0; nt < 4; ++nt)
#pragma unroll
                for (int j = 0; j < 8; ++j)
                    acc[mt][nt][j] = 0.0f;

        for (int kk = 0; kk < DIN; kk += BK) {
            const int p = s & 1;
            WAIT_ASYNC0();       // this wave's stage-s copies have landed in LDS
            __syncthreads();     // all waves landed; buffer 1-p free for reuse

            if (kk + BK < DIN)       issueStage(e, kk + BK, p ^ 1);
            else if (e + 1 < NEXP)   issueStage(e + 1, 0, p ^ 1);

            // ---- two k-steps of 32 from buffer p
#pragma unroll
            for (int ks = 0; ks < BK; ks += 32) {
                v16bf af[2];
#pragma unroll
                for (int mt = 0; mt < 2; ++mt) {
                    // A frag: lanes<16 K{0..7,16..23}, lanes>=16 K{8..15,24..31}
                    const __bf16* ap = &A2[p][(wm * 32 + mt * 16 + ln15) * LDT + ks + hi * 8];
                    v8bf lo = *(const v8bf*)ap;
                    v8bf hb = *(const v8bf*)(ap + 16);
#pragma unroll
                    for (int q = 0; q < 8; ++q) { af[mt][q] = lo[q]; af[mt][q + 8] = hb[q]; }
                }
                v16bf bf4[4];
#pragma unroll
                for (int nt = 0; nt < 4; ++nt) {
                    // B frag: lanes<16 K{0..7,16..23}, lanes>=16 K{8..15,24..31}
                    const __bf16* bp = &B2[p][(wn * 64 + nt * 16 + ln15) * LDT + ks + hi * 8];
                    v8bf lo = *(const v8bf*)bp;
                    v8bf hb = *(const v8bf*)(bp + 16);
#pragma unroll
                    for (int q = 0; q < 8; ++q) { bf4[nt][q] = lo[q]; bf4[nt][q + 8] = hb[q]; }
                }
#pragma unroll
                for (int mt = 0; mt < 2; ++mt)
#pragma unroll
                    for (int nt = 0; nt < 4; ++nt)
                        acc[mt][nt] = __builtin_amdgcn_wmma_f32_16x16x32_bf16(
                            false, af[mt], false, bf4[nt],
                            (short)0, acc[mt][nt], false, false);
            }
            ++s;
        }

        // ---- fuse: combined += gate[row,e] * exp(acc + bias[e,col])
#pragma unroll
        for (int mt = 0; mt < 2; ++mt) {
            float gv[8];
#pragma unroll
            for (int j = 0; j < 8; ++j) {
                int row = rowBase + wm * 32 + mt * 16 + j + hi * 8;
                gv[j] = gates[(size_t)row * NEXP + e];
            }
#pragma unroll
            for (int nt = 0; nt < 4; ++nt) {
                int col = colBase + wn * 64 + nt * 16 + ln15;
                float bv = bias[e * DOUT + col];
#pragma unroll
                for (int j = 0; j < 8; ++j) {
                    float x = acc[mt][nt][j] + bv;
                    comb[mt][nt][j] += (gv[j] != 0.0f) ? gv[j] * __expf(x) : 0.0f;
                }
            }
        }
    }

#pragma unroll
    for (int mt = 0; mt < 2; ++mt) {
#pragma unroll
        for (int nt = 0; nt < 4; ++nt) {
            int col = colBase + wn * 64 + nt * 16 + ln15;
#pragma unroll
            for (int j = 0; j < 8; ++j) {
                int row = rowBase + wm * 32 + mt * 16 + j + hi * 8;
                float c = comb[mt][nt][j];
                c = (c == 0.0f) ? 2.2204460492503131e-16f : c;
                out[(size_t)row * DOUT + col] = __logf(c);
            }
        }
    }
}

// --------------- fallback: fused-convert version (no workspace) ---------------
__global__ __launch_bounds__(256)
void moe_logexp_wmma_f32src(const float* __restrict__ gates, const float* __restrict__ inp,
                            const float* __restrict__ W, const float* __restrict__ bias,
                            float* __restrict__ out)
{
    __shared__ __align__(16) __bf16 A_s[BM * 520];
    __shared__ __align__(16) __bf16 B_s[BN * LDT];
    const int tid = threadIdx.x, lane = tid & 31, wv = tid >> 5;
    const int wm = wv >> 1, wn = wv & 1, ln15 = lane & 15, hi = lane >> 4;
    const int rowBase = (blockIdx.x >> 2) * BM, colBase = (blockIdx.x & 3) * BN;
    {
        const float* abase = inp + (size_t)rowBase * DIN;
#pragma unroll 4
        for (int i = 0; i < 64; ++i) {
            int linear = tid + i * 256, row = linear >> 7, c4 = (linear & 127) << 2;
            float4 v = *(const float4*)(abase + (size_t)row * DIN + c4);
            __bf16* dst = &A_s[row * 520 + c4];
            dst[0] = (__bf16)v.x; dst[1] = (__bf16)v.y; dst[2] = (__bf16)v.z; dst[3] = (__bf16)v.w;
        }
    }
    float4 wreg[8];
    auto loadBregs = [&](int e, int kk) {
        const float* wb = W + ((size_t)e * DIN + kk) * DOUT + colBase;
#pragma unroll
        for (int i = 0; i < 8; ++i) {
            int linear = tid + i * 256, k = linear >> 5, n = (linear & 31) << 2;
            wreg[i] = *(const float4*)(wb + (size_t)k * DOUT + n);
        }
    };
    v8f comb[2][4];
#pragma unroll
    for (int mt = 0; mt < 2; ++mt)
#pragma unroll
        for (int nt = 0; nt < 4; ++nt)
#pragma unroll
            for (int j = 0; j < 8; ++j) comb[mt][nt][j] = 0.0f;
    loadBregs(0, 0);
    for (int e = 0; e < NEXP; ++e) {
        v8f acc[2][4];
#pragma unroll
        for (int mt = 0; mt < 2; ++mt)
#pragma unroll
            for (int nt = 0; nt < 4; ++nt)
#pragma unroll
                for (int j = 0; j < 8; ++j) acc[mt][nt][j] = 0.0f;
        for (int kk = 0; kk < DIN; kk += BK) {
            __syncthreads();
#pragma unroll
            for (int i = 0; i < 8; ++i) {
                int linear = tid + i * 256, k = linear >> 5, n = (linear & 31) << 2;
                float4 v = wreg[i];
                B_s[(n + 0) * LDT + k] = (__bf16)v.x; B_s[(n + 1) * LDT + k] = (__bf16)v.y;
                B_s[(n + 2) * LDT + k] = (__bf16)v.z; B_s[(n + 3) * LDT + k] = (__bf16)v.w;
            }
            __syncthreads();
            if (kk + BK < DIN)     loadBregs(e, kk + BK);
            else if (e + 1 < NEXP) loadBregs(e + 1, 0);
#pragma unroll
            for (int ks = 0; ks < BK; ks += 32) {
                v16bf af[2];
#pragma unroll
                for (int mt = 0; mt < 2; ++mt) {
                    const __bf16* ap = &A_s[(wm * 32 + mt * 16 + ln15) * 520 + kk + ks + hi * 8];
                    v8bf lo = *(const v8bf*)ap, hb = *(const v8bf*)(ap + 16);
#pragma unroll
                    for (int q = 0; q < 8; ++q) { af[mt][q] = lo[q]; af[mt][q + 8] = hb[q]; }
                }
                v16bf bf4[4];
#pragma unroll
                for (int nt = 0; nt < 4; ++nt) {
                    const __bf16* bp = &B_s[(wn * 64 + nt * 16 + ln15) * LDT + ks + hi * 8];
                    v8bf lo = *(const v8bf*)bp, hb = *(const v8bf*)(bp + 16);
#pragma unroll
                    for (int q = 0; q < 8; ++q) { bf4[nt][q] = lo[q]; bf4[nt][q + 8] = hb[q]; }
                }
#pragma unroll
                for (int mt = 0; mt < 2; ++mt)
#pragma unroll
                    for (int nt = 0; nt < 4; ++nt)
                        acc[mt][nt] = __builtin_amdgcn_wmma_f32_16x16x32_bf16(
                            false, af[mt], false, bf4[nt], (short)0, acc[mt][nt], false, false);
            }
        }
#pragma unroll
        for (int mt = 0; mt < 2; ++mt) {
            float gv[8];
#pragma unroll
            for (int j = 0; j < 8; ++j)
                gv[j] = gates[(size_t)(rowBase + wm * 32 + mt * 16 + j + hi * 8) * NEXP + e];
#pragma unroll
            for (int nt = 0; nt < 4; ++nt) {
                int col = colBase + wn * 64 + nt * 16 + ln15;
                float bv = bias[e * DOUT + col];
#pragma unroll
                for (int j = 0; j < 8; ++j) {
                    float x = acc[mt][nt][j] + bv;
                    comb[mt][nt][j] += (gv[j] != 0.0f) ? gv[j] * __expf(x) : 0.0f;
                }
            }
        }
    }
#pragma unroll
    for (int mt = 0; mt < 2; ++mt)
#pragma unroll
        for (int nt = 0; nt < 4; ++nt) {
            int col = colBase + wn * 64 + nt * 16 + ln15;
#pragma unroll
            for (int j = 0; j < 8; ++j) {
                int row = rowBase + wm * 32 + mt * 16 + j + hi * 8;
                float c = comb[mt][nt][j];
                c = (c == 0.0f) ? 2.2204460492503131e-16f : c;
                out[(size_t)row * DOUT + col] = __logf(c);
            }
        }
}

extern "C" void kernel_launch(void* const* d_in, const int* in_sizes, int n_in,
                              void* d_out, int out_size, void* d_ws, size_t ws_size,
                              hipStream_t stream) {
    const float* gates = (const float*)d_in[0];   // [32768, 8]
    const float* inp   = (const float*)d_in[1];   // [32768, 512]
    const float* W     = (const float*)d_in[2];   // [8, 512, 512]
    const float* bias  = (const float*)d_in[3];   // [8, 512]
    float* out = (float*)d_out;                   // [32768, 512]

    const int B = 32768;
    const size_t needW = (size_t)NEXP * DIN * DOUT * 2;      // 4 MB
    const size_t needI = (size_t)B * DIN * 2;                // 32 MB
    dim3 grid((B / BM) * (DOUT / BN)), block(256);

    if (ws_size >= needW + needI) {
        __bf16* Wt     = (__bf16*)d_ws;
        __bf16* inp_bf = (__bf16*)((char*)d_ws + needW);
        cvt_w_transpose_kernel<<<dim3(NEXP * 64), block, 0, stream>>>(W, Wt);
        cvt_inp_kernel<<<dim3((B * DIN) / (256 * 8)), block, 0, stream>>>(inp, inp_bf);
        moe_logexp_wmma_bf16<<<grid, block, 0, stream>>>(gates, inp_bf, Wt, bias, out);
    } else {
        moe_logexp_wmma_f32src<<<grid, block, 0, stream>>>(gates, inp, W, bias, out);
    }
}
